// GPTDecoderBlock_73272142069952
// MI455X (gfx1250) — matrix operand order repaired and checked by reading, not compiled, not run on hardware
//
#include <hip/hip_runtime.h>

// ---------------------------------------------------------------------------
// Types for CDNA5 WMMA (gfx1250, wave32)
// ---------------------------------------------------------------------------
typedef __bf16 v16bf __attribute__((ext_vector_type(16)));
typedef __bf16 v4bf  __attribute__((ext_vector_type(4)));
typedef float  v8f   __attribute__((ext_vector_type(8)));

#define Bb 2
#define Ss 1024
#define Dd 2048
#define Hh 16
#define DH 128
#define Ff 8192
#define TOK (Bb * Ss)           // 2048 token rows
#define CN_SCALE (2048.0f / 2047.0f)
#define MAX_ALPHA 0.2f

static __device__ __forceinline__ v8f wmma_bf16(v16bf a, v16bf b, v8f c) {
  return __builtin_amdgcn_wmma_f32_16x16x32_bf16(
      /*neg_a=*/false, a, /*neg_b=*/false, b,
      /*c_mod=*/(short)0, c, /*reuse_a=*/false, /*reuse_b=*/false);
}

// ---- CDNA5 async global->LDS copy (16B per lane), with safe fallback ------
#if __has_builtin(__builtin_amdgcn_global_load_async_to_lds_b128)
#define HAVE_ASYNC_LDS 1
typedef int i32x4 __attribute__((vector_size(16)));
typedef __attribute__((address_space(1))) i32x4* as1_i32x4_t;
typedef __attribute__((address_space(3))) i32x4* as3_i32x4_t;
static __device__ __forceinline__ void cp16_g2l(__bf16* lds, const __bf16* g) {
  __builtin_amdgcn_global_load_async_to_lds_b128(
      (as1_i32x4_t)g, (as3_i32x4_t)lds, 0, 0);
}
#else
#define HAVE_ASYNC_LDS 0
static __device__ __forceinline__ void cp16_g2l(__bf16* lds, const __bf16* g) {
  *(uint4*)lds = *(const uint4*)g;
}
#endif

static __device__ __forceinline__ void wait_async_lds() {
#if HAVE_ASYNC_LDS
#if __has_builtin(__builtin_amdgcn_s_wait_asynccnt)
  __builtin_amdgcn_s_wait_asynccnt(0);
#else
  asm volatile("s_wait_asynccnt 0x0" ::: "memory");
#endif
#endif
}

// ---------------------------------------------------------------------------
// fp32 -> bf16 bulk conversion (4 elements / thread)
// ---------------------------------------------------------------------------
__global__ __launch_bounds__(256)
void cvt_bf16_kernel(const float* __restrict__ in, __bf16* __restrict__ out, int n4) {
  int i = blockIdx.x * 256 + threadIdx.x;
  if (i < n4) {
    float4 v = ((const float4*)in)[i];
    v4bf o;
    o[0] = (__bf16)v.x; o[1] = (__bf16)v.y; o[2] = (__bf16)v.z; o[3] = (__bf16)v.w;
    *(v4bf*)(out + 4 * (size_t)i) = o;
  }
}

// ---------------------------------------------------------------------------
// CenterNorm: y = g * (scale * (x - mean(x))) + b ; also emits bf16 copy
// ---------------------------------------------------------------------------
__global__ __launch_bounds__(256)
void centernorm_kernel(const float* __restrict__ x, const float* __restrict__ g,
                       const float* __restrict__ bta, float* __restrict__ y,
                       __bf16* __restrict__ ybf) {
  __shared__ float red[256];
  const int row = blockIdx.x;
  const int t = threadIdx.x;
  const float* xr = x + (size_t)row * Dd;
  float s = 0.f;
  for (int i = t; i < Dd; i += 256) s += xr[i];
  red[t] = s;
  __syncthreads();
  for (int off = 128; off; off >>= 1) {
    if (t < off) red[t] += red[t + off];
    __syncthreads();
  }
  const float mu = red[0] * (1.0f / (float)Dd);
  float* yr = y + (size_t)row * Dd;
  __bf16* yb = ybf + (size_t)row * Dd;
  for (int i = t; i < Dd; i += 256) {
    float v = g[i] * (CN_SCALE * (xr[i] - mu)) + bta[i];
    yr[i] = v;
    yb[i] = (__bf16)v;
  }
}

// ---------------------------------------------------------------------------
// bf16 WMMA GEMM:  C[M,N] = epilogue( A[M,K] * Bw[N,K]^T + bias[N] )
// A, Bw pre-converted bf16. Block tile 128x64x64; 256 thr = 8 waves (32x32 ea).
// EPI: 0 = bias only, 1 = tanh-GELU(bias), 2 = res + clip(alpha)*(acc+bias)
// OUTBF: 1 -> write bf16 to Cb, else fp32 to Cf.
// ---------------------------------------------------------------------------
template <int EPI, int OUTBF>
__global__ __launch_bounds__(256)
void gemm_bf16_kernel(const __bf16* __restrict__ A, const __bf16* __restrict__ Bw,
                      const float* __restrict__ bias, const float* __restrict__ res,
                      const float* __restrict__ alpha, float* __restrict__ Cf,
                      __bf16* __restrict__ Cb, int M, int N, int K) {
  __shared__ __bf16 As[128 * 64];
  __shared__ __bf16 Bs[64 * 64];

  const int tid = threadIdx.x;
  const int wave = tid >> 5;
  const int lane = tid & 31;
  const int ln16 = lane & 15;
  const int hi   = (lane >= 16) ? 1 : 0;
  const int wave_m = wave >> 1;      // 0..3
  const int wave_n = wave & 1;       // 0..1
  const int rowBase = blockIdx.y * 128;
  const int colBase = blockIdx.x * 64;

  v8f acc[2][2];
#pragma unroll
  for (int i = 0; i < 2; ++i)
#pragma unroll
    for (int j = 0; j < 2; ++j)
#pragma unroll
      for (int r = 0; r < 8; ++r) acc[i][j][r] = 0.f;

  for (int k0 = 0; k0 < K; k0 += 64) {
    __syncthreads();
    // A tile: 128 rows x 64 cols bf16 = 1024 x 16B chunks, 4 per thread
#pragma unroll
    for (int it = 0; it < 4; ++it) {
      int slot = tid + it * 256;
      int row = slot >> 3;
      int c8 = (slot & 7) << 3;
      cp16_g2l(&As[row * 64 + c8], A + (size_t)(rowBase + row) * K + k0 + c8);
    }
    // B tile: 64 rows x 64 cols bf16 = 512 x 16B chunks, 2 per thread
#pragma unroll
    for (int it = 0; it < 2; ++it) {
      int slot = tid + it * 256;
      int row = slot >> 3;
      int c8 = (slot & 7) << 3;
      cp16_g2l(&Bs[row * 64 + c8], Bw + (size_t)(colBase + row) * K + k0 + c8);
    }
    // prefetch next K tile into L2 while WMMAs run
    if (k0 + 64 < K) {
      __builtin_prefetch(A + (size_t)(rowBase + (tid >> 1)) * K + k0 + 64 + (tid & 1) * 32, 0, 1);
      if (tid < 128)
        __builtin_prefetch(Bw + (size_t)(colBase + (tid >> 1)) * K + k0 + 64 + (tid & 1) * 32, 0, 1);
    }
    wait_async_lds();
    __syncthreads();

#pragma unroll
    for (int kk = 0; kk < 64; kk += 32) {
      v16bf af[2], bf_[2];
#pragma unroll
      for (int tm = 0; tm < 2; ++tm) {
        int m_local = wave_m * 32 + tm * 16 + ln16;
#pragma unroll
        for (int i = 0; i < 16; ++i) {
          int kidx = kk + i + (hi ? 16 : 0);
          af[tm][i] = As[m_local * 64 + kidx];
        }
      }
#pragma unroll
      for (int tn = 0; tn < 2; ++tn) {
        int n_local = wave_n * 32 + tn * 16 + ln16;
#pragma unroll
        for (int i = 0; i < 16; ++i) {
          int kidx = kk + i + (hi ? 16 : 0);
          bf_[tn][i] = Bs[n_local * 64 + kidx];
        }
      }
#pragma unroll
      for (int tm = 0; tm < 2; ++tm)
#pragma unroll
        for (int tn = 0; tn < 2; ++tn)
          acc[tm][tn] = wmma_bf16(af[tm], bf_[tn], acc[tm][tn]);
    }
  }

  // Epilogue: C/D layout m = r + 8*hi, n = lane%16
#pragma unroll
  for (int tm = 0; tm < 2; ++tm) {
#pragma unroll
    for (int tn = 0; tn < 2; ++tn) {
#pragma unroll
      for (int r = 0; r < 8; ++r) {
        int row = rowBase + wave_m * 32 + tm * 16 + r + 8 * hi;
        int col = colBase + wave_n * 32 + tn * 16 + ln16;
        float v = acc[tm][tn][r] + bias[col];
        if (EPI == 1) {
          float u = 0.7978845608028654f * (v + 0.044715f * v * v * v);
          v = 0.5f * v * (1.0f + tanhf(u));
        } else if (EPI == 2) {
          float a = alpha[col];
          a = fminf(fmaxf(a, -MAX_ALPHA), MAX_ALPHA);
          v = res[(size_t)row * N + col] + a * v;
        }
        if (OUTBF)
          Cb[(size_t)row * N + col] = (__bf16)v;
        else
          Cf[(size_t)row * N + col] = v;
      }
    }
  }
}

// ---------------------------------------------------------------------------
// Cosine-similarity causal flash attention.
// Grid: (S/64, B*H). Block: 128 threads = 4 waves; each wave owns 16 queries.
// q/k/v fp32 [B*S, D] head-major cols; output written as bf16 [B*S, D].
// ---------------------------------------------------------------------------
__global__ __launch_bounds__(128)
void attn_kernel(const float* __restrict__ q, const float* __restrict__ k,
                 const float* __restrict__ v, const float* __restrict__ tau,
                 const float* __restrict__ nu, __bf16* __restrict__ o) {
  __shared__ __bf16 qs[64 * 128];   // normalized q rows [qrow][d]
  __shared__ __bf16 ks[64 * 128];   // normalized k rows [key][d]
  __shared__ __bf16 vsT[128 * 64];  // v transposed [d][key]
  __shared__ __bf16 ps[64 * 64];    // probabilities [qrow][key]

  const int tid = threadIdx.x;
  const int wave = tid >> 5;
  const int lane = tid & 31;
  const int ln16 = lane & 15;
  const int hi = (lane >= 16) ? 1 : 0;
  const int bh = blockIdx.y;
  const int b = bh >> 4;
  const int h = bh & 15;
  const int qb = blockIdx.x * 64;

  const float tauh = tau[h];
  const float nuh = nu[h];

  // --- load + l2-normalize this wave's 16 query rows (2 lanes per row) ---
  {
    int r = lane >> 1, half = lane & 1;
    int qrow = wave * 16 + r;
    const float* src = q + ((size_t)(b * Ss + qb + qrow)) * Dd + h * DH + half * 64;
    float ss = 0.f;
#pragma unroll
    for (int j = 0; j < 16; ++j) {
      float4 t4 = ((const float4*)src)[j];
      ss += t4.x * t4.x + t4.y * t4.y + t4.z * t4.z + t4.w * t4.w;
    }
    ss += __shfl_xor(ss, 1, 32);
    float inv = 1.0f / (sqrtf(ss) + 1e-6f);
#pragma unroll
    for (int j = 0; j < 16; ++j) {
      float4 t4 = ((const float4*)src)[j];
      int d0 = half * 64 + j * 4;
      qs[qrow * 128 + d0 + 0] = (__bf16)(t4.x * inv);
      qs[qrow * 128 + d0 + 1] = (__bf16)(t4.y * inv);
      qs[qrow * 128 + d0 + 2] = (__bf16)(t4.z * inv);
      qs[qrow * 128 + d0 + 3] = (__bf16)(t4.w * inv);
    }
  }

  v8f acc_o[8];
#pragma unroll
  for (int dt = 0; dt < 8; ++dt)
#pragma unroll
    for (int r = 0; r < 8; ++r) acc_o[dt][r] = 0.f;
  float mrow[8], lrow[8];
#pragma unroll
  for (int r = 0; r < 8; ++r) { mrow[r] = -3.0e38f; lrow[r] = 0.f; }

  const int nkb = blockIdx.x + 1;
  for (int kb = 0; kb < nkb; ++kb) {
    __syncthreads();
    // --- cooperative load: normalized K rows + transposed V (64 keys) ---
    {
      int r = tid >> 1, half = tid & 1;
      const float* ksrc = k + ((size_t)(b * Ss + kb * 64 + r)) * Dd + h * DH + half * 64;
      float ss = 0.f;
#pragma unroll
      for (int j = 0; j < 16; ++j) {
        float4 t4 = ((const float4*)ksrc)[j];
        ss += t4.x * t4.x + t4.y * t4.y + t4.z * t4.z + t4.w * t4.w;
      }
      ss += __shfl_xor(ss, 1, 32);
      float inv = 1.0f / (sqrtf(ss) + 1e-6f);
#pragma unroll
      for (int j = 0; j < 16; ++j) {
        float4 t4 = ((const float4*)ksrc)[j];
        int d0 = half * 64 + j * 4;
        ks[r * 128 + d0 + 0] = (__bf16)(t4.x * inv);
        ks[r * 128 + d0 + 1] = (__bf16)(t4.y * inv);
        ks[r * 128 + d0 + 2] = (__bf16)(t4.z * inv);
        ks[r * 128 + d0 + 3] = (__bf16)(t4.w * inv);
      }
      const float* vsrc = v + ((size_t)(b * Ss + kb * 64 + r)) * Dd + h * DH + half * 64;
#pragma unroll
      for (int j = 0; j < 16; ++j) {
        float4 t4 = ((const float4*)vsrc)[j];
        int d0 = half * 64 + j * 4;
        vsT[(d0 + 0) * 64 + r] = (__bf16)t4.x;
        vsT[(d0 + 1) * 64 + r] = (__bf16)t4.y;
        vsT[(d0 + 2) * 64 + r] = (__bf16)t4.z;
        vsT[(d0 + 3) * 64 + r] = (__bf16)t4.w;
      }
    }
    __syncthreads();

    // --- scores: 16 queries x 64 keys via WMMA over DH=128 ---
    v8f sacc[4];
#pragma unroll
    for (int tn = 0; tn < 4; ++tn)
#pragma unroll
      for (int r = 0; r < 8; ++r) sacc[tn][r] = 0.f;

#pragma unroll
    for (int kk = 0; kk < 4; ++kk) {
      v16bf af;
      int m_local = wave * 16 + ln16;
#pragma unroll
      for (int i = 0; i < 16; ++i) {
        int kidx = kk * 32 + i + (hi ? 16 : 0);
        af[i] = qs[m_local * 128 + kidx];
      }
      // pre-build all 4 B fragments in distinct regs so ds_loads overlap WMMA
      v16bf bf4[4];
#pragma unroll
      for (int tn = 0; tn < 4; ++tn) {
        int key = tn * 16 + ln16;
#pragma unroll
        for (int i = 0; i < 16; ++i) {
          int kidx = kk * 32 + i + (hi ? 16 : 0);
          bf4[tn][i] = ks[key * 128 + kidx];
        }
      }
#pragma unroll
      for (int tn = 0; tn < 4; ++tn)
        sacc[tn] = wmma_bf16(af, bf4[tn], sacc[tn]);
    }

    // --- online softmax per row (rows live in 16-lane halves) ---
    float scale_[8];
#pragma unroll
    for (int r = 0; r < 8; ++r) {
      int m = r + 8 * hi;
      int qg = qb + wave * 16 + m;
      float srow[4];
      float sm = -3.0e38f;
#pragma unroll
      for (int tn = 0; tn < 4; ++tn) {
        int kg = kb * 64 + tn * 16 + ln16;
        float sv = sacc[tn][r] * tauh;
        if (kg > qg) sv = -3.0e38f;
        srow[tn] = sv;
        sm = fmaxf(sm, sv);
      }
#pragma unroll
      for (int msk = 8; msk; msk >>= 1) sm = fmaxf(sm, __shfl_xor(sm, msk, 32));
      float mnew = fmaxf(mrow[r], sm);
      float sc = __expf(mrow[r] - mnew);
      mrow[r] = mnew;
      float psum = 0.f;
#pragma unroll
      for (int tn = 0; tn < 4; ++tn) {
        float p = __expf(srow[tn] - mnew);
        psum += p;
        ps[(wave * 16 + m) * 64 + tn * 16 + ln16] = (__bf16)p;
      }
#pragma unroll
      for (int msk = 8; msk; msk >>= 1) psum += __shfl_xor(psum, msk, 32);
      lrow[r] = lrow[r] * sc + psum;
      scale_[r] = sc;
    }
#pragma unroll
    for (int dt = 0; dt < 8; ++dt)
#pragma unroll
      for (int r = 0; r < 8; ++r) acc_o[dt][r] *= scale_[r];

    // --- P(16x64) x V(64x128) via WMMA, d-tiles in groups of 4 ---
#pragma unroll
    for (int kk2 = 0; kk2 < 2; ++kk2) {
      v16bf pf;
#pragma unroll
      for (int i = 0; i < 16; ++i) {
        int kidx = kk2 * 32 + i + (hi ? 16 : 0);
        pf[i] = ps[(wave * 16 + ln16) * 64 + kidx];
      }
#pragma unroll
      for (int dg = 0; dg < 2; ++dg) {
        v16bf vf4[4];
#pragma unroll
        for (int dj = 0; dj < 4; ++dj) {
          int dt = dg * 4 + dj;
#pragma unroll
          for (int i = 0; i < 16; ++i) {
            int key = kk2 * 32 + i + (hi ? 16 : 0);
            vf4[dj][i] = vsT[(dt * 16 + ln16) * 64 + key];
          }
        }
#pragma unroll
        for (int dj = 0; dj < 4; ++dj)
          acc_o[dg * 4 + dj] = wmma_bf16(pf, vf4[dj], acc_o[dg * 4 + dj]);
      }
    }
  }

  // --- write out (bf16): o[b,s, h*DH + d] = acc * nu / l ---
#pragma unroll
  for (int dt = 0; dt < 8; ++dt) {
#pragma unroll
    for (int r = 0; r < 8; ++r) {
      int m = r + 8 * hi;
      int qg = qb + wave * 16 + m;
      int dcol = dt * 16 + ln16;
      float val = acc_o[dt][r] * nuh / lrow[r];
      o[((size_t)(b * Ss + qg)) * Dd + h * DH + dcol] = (__bf16)val;
    }
  }
}

// ---------------------------------------------------------------------------
// Host orchestration
// ---------------------------------------------------------------------------
extern "C" void kernel_launch(void* const* d_in, const int* in_sizes, int n_in,
                              void* d_out, int out_size, void* d_ws, size_t ws_size,
                              hipStream_t stream) {
  (void)in_sizes; (void)n_in; (void)out_size; (void)ws_size;
  const float* x      = (const float*)d_in[0];
  const float* ln1_g  = (const float*)d_in[1];
  const float* ln1_b  = (const float*)d_in[2];
  const float* wq     = (const float*)d_in[3];
  const float* bq     = (const float*)d_in[4];
  const float* wk     = (const float*)d_in[5];
  const float* bk     = (const float*)d_in[6];
  const float* wv     = (const float*)d_in[7];
  const float* bv     = (const float*)d_in[8];
  const float* wo     = (const float*)d_in[9];
  const float* bo     = (const float*)d_in[10];
  const float* tau    = (const float*)d_in[11];
  const float* nu     = (const float*)d_in[12];
  const float* alpha1 = (const float*)d_in[13];
  const float* ln2_g  = (const float*)d_in[14];
  const float* ln2_b  = (const float*)d_in[15];
  const float* w1     = (const float*)d_in[16];
  const float* b1     = (const float*)d_in[17];
  const float* w2     = (const float*)d_in[18];
  const float* b2     = (const float*)d_in[19];
  const float* alpha2 = (const float*)d_in[20];
  float* out = (float*)d_out;

  // workspace carve-up (256B aligned chunks)
  char* wsp = (char*)d_ws;
  auto alloc = [&](size_t bytes) {
    char* p = wsp;
    wsp += (bytes + 255) & ~(size_t)255;
    return p;
  };
  const size_t TDf = (size_t)TOK * Dd * sizeof(float);   // 16 MB
  const size_t TDb = (size_t)TOK * Dd * sizeof(__bf16);  // 8 MB
  const size_t DDb = (size_t)Dd * Dd * sizeof(__bf16);   // 8 MB
  const size_t DFb = (size_t)Dd * Ff * sizeof(__bf16);   // 32 MB

  float*  a_buf  = (float*)alloc(TDf);
  __bf16* a_bf   = (__bf16*)alloc(TDb);
  float*  q_buf  = (float*)alloc(TDf);
  float*  k_buf  = (float*)alloc(TDf);
  float*  v_buf  = (float*)alloc(TDf);
  __bf16* o_bf   = (__bf16*)alloc(TDb);
  float*  x1     = (float*)alloc(TDf);
  float*  m_buf  = (float*)alloc(TDf);
  __bf16* m_bf   = (__bf16*)alloc(TDb);
  __bf16* h_bf   = (__bf16*)alloc((size_t)TOK * Ff * sizeof(__bf16));
  __bf16* wq_bf  = (__bf16*)alloc(DDb);
  __bf16* wk_bf  = (__bf16*)alloc(DDb);
  __bf16* wv_bf  = (__bf16*)alloc(DDb);
  __bf16* wo_bf  = (__bf16*)alloc(DDb);
  __bf16* w1_bf  = (__bf16*)alloc(DFb);
  __bf16* w2_bf  = (__bf16*)alloc(DFb);

  // 0) one-shot weight conversion to bf16 (stays hot in 192MB L2)
  const int nDD4 = Dd * Dd / 4, nDF4 = Dd * Ff / 4;
  cvt_bf16_kernel<<<nDD4 / 256, 256, 0, stream>>>(wq, wq_bf, nDD4);
  cvt_bf16_kernel<<<nDD4 / 256, 256, 0, stream>>>(wk, wk_bf, nDD4);
  cvt_bf16_kernel<<<nDD4 / 256, 256, 0, stream>>>(wv, wv_bf, nDD4);
  cvt_bf16_kernel<<<nDD4 / 256, 256, 0, stream>>>(wo, wo_bf, nDD4);
  cvt_bf16_kernel<<<nDF4 / 256, 256, 0, stream>>>(w1, w1_bf, nDF4);
  cvt_bf16_kernel<<<nDF4 / 256, 256, 0, stream>>>(w2, w2_bf, nDF4);

  // 1) a = CenterNorm(x)  (fp32 + bf16)
  centernorm_kernel<<<TOK, 256, 0, stream>>>(x, ln1_g, ln1_b, a_buf, a_bf);

  // 2) Q/K/V projections (fp32 out for normalization precision)
  dim3 gproj(Dd / 64, TOK / 128);
  gemm_bf16_kernel<0, 0><<<gproj, 256, 0, stream>>>(a_bf, wq_bf, bq, nullptr, nullptr, q_buf, nullptr, TOK, Dd, Dd);
  gemm_bf16_kernel<0, 0><<<gproj, 256, 0, stream>>>(a_bf, wk_bf, bk, nullptr, nullptr, k_buf, nullptr, TOK, Dd, Dd);
  gemm_bf16_kernel<0, 0><<<gproj, 256, 0, stream>>>(a_bf, wv_bf, bv, nullptr, nullptr, v_buf, nullptr, TOK, Dd, Dd);

  // 3) cosine causal attention -> bf16 output
  attn_kernel<<<dim3(Ss / 64, Bb * Hh), 128, 0, stream>>>(q_buf, k_buf, v_buf, tau, nu, o_bf);

  // 4) x1 = a + clip(alpha1) * (o @ wo^T + bo)
  gemm_bf16_kernel<2, 0><<<gproj, 256, 0, stream>>>(o_bf, wo_bf, bo, a_buf, alpha1, x1, nullptr, TOK, Dd, Dd);

  // 5) m = CenterNorm(x1)  (fp32 + bf16)
  centernorm_kernel<<<TOK, 256, 0, stream>>>(x1, ln2_g, ln2_b, m_buf, m_bf);

  // 6) h = gelu(m @ w1^T + b1) -> bf16
  gemm_bf16_kernel<1, 1><<<dim3(Ff / 64, TOK / 128), 256, 0, stream>>>(m_bf, w1_bf, b1, nullptr, nullptr, nullptr, h_bf, TOK, Ff, Dd);

  // 7) out = m + clip(alpha2) * (h @ w2^T + b2)
  gemm_bf16_kernel<2, 0><<<gproj, 256, 0, stream>>>(h_bf, w2_bf, b2, m_buf, alpha2, out, nullptr, TOK, Dd, Ff);
}
